// Hamming_Embedding_45251775430700
// MI455X (gfx1250) — 8-bit matrix operand order repaired and checked by reading, compiled, not run on hardware
//
#include <hip/hip_runtime.h>
#include <stdint.h>

typedef int v8i __attribute__((ext_vector_type(8)));

#define BATCH   4096
#define D_MODEL 1024
#define L_DIM   2048

__global__ void __launch_bounds__(256)
pack_kernel(const int* __restrict__ x, const float* __restrict__ refs,
            unsigned char* __restrict__ xb, unsigned char* __restrict__ rb,
            float* __restrict__ sum_x, float* __restrict__ sum_r) {
    const int blk = blockIdx.x;
    const int t   = threadIdx.x;
    int s = 0;

    if (blk < BATCH) {
        const int row = blk;
        const int4* xr = (const int4*)(x + (size_t)row * L_DIM);
        uint* out = (uint*)(xb + (size_t)row * L_DIM);
#pragma unroll
        for (int i = 0; i < 2; ++i) {
            const int idx = t + i * 256;
            int4 v = xr[idx];
            s += v.x + v.y + v.z + v.w;
            uint packed = (uint)(v.x & 0xFF) | ((uint)(v.y & 0xFF) << 8) |
                          ((uint)(v.z & 0xFF) << 16) | ((uint)(v.w & 0xFF) << 24);
            out[idx] = packed;
        }
    } else {
        const int row = blk - BATCH;
        const float4* rr = (const float4*)(refs + (size_t)row * L_DIM);
        uint* out = (uint*)(rb + (size_t)row * L_DIM);
#pragma unroll
        for (int i = 0; i < 2; ++i) {
            const int idx = t + i * 256;
            float4 v = rr[idx];
            int a0 = (int)v.x, a1 = (int)v.y, a2 = (int)v.z, a3 = (int)v.w;
            s += a0 + a1 + a2 + a3;
            uint packed = (uint)(a0 & 0xFF) | ((uint)(a1 & 0xFF) << 8) |
                          ((uint)(a2 & 0xFF) << 16) | ((uint)(a3 & 0xFF) << 24);
            out[idx] = packed;
        }
    }

    __shared__ int red[256];
    red[t] = s;
    __syncthreads();
    for (int off = 128; off > 0; off >>= 1) {
        if (t < off) red[t] += red[t + off];
        __syncthreads();
    }
    if (t == 0) {
        if (blk < BATCH) sum_x[blk] = (float)red[0];
        else             sum_r[blk - BATCH] = (float)red[0];
    }
}

__device__ __forceinline__ v8i load_a_frag(const unsigned char* p) {
    int4 d0 = *(const int4*)(p);
    int4 d1 = *(const int4*)(p + 32);
    v8i a;
    a[0] = d0.x; a[1] = d0.y; a[2] = d0.z; a[3] = d0.w;
    a[4] = d1.x; a[5] = d1.y; a[6] = d1.z; a[7] = d1.w;
    return a;
}

__device__ __forceinline__ v8i load_b_frag(const unsigned char* p) {
    int4 d0 = *(const int4*)(p);
    int4 d1 = *(const int4*)(p + 32);
    v8i b;
    b[0] = d0.x; b[1] = d0.y; b[2] = d0.z; b[3] = d0.w;
    b[4] = d1.x; b[5] = d1.y; b[6] = d1.z; b[7] = d1.w;
    return b;
}

__global__ void __launch_bounds__(256)
hamming_wmma_kernel(const unsigned char* __restrict__ xb,
                    const unsigned char* __restrict__ rb,
                    const float* __restrict__ sum_x,
                    const float* __restrict__ sum_r,
                    float* __restrict__ out) {
    const int lane = threadIdx.x & 31;
    const int wave = threadIdx.x >> 5;
    const int wm   = wave & 3;
    const int wn   = wave >> 2;
    const int half = lane >> 4;
    const int l16  = lane & 15;

    const int m_base = blockIdx.x * 128 + wm * 32;
    const int n_base = blockIdx.y * 128 + wn * 64;

    v8i acc[2][4];
#pragma unroll
    for (int mi = 0; mi < 2; ++mi)
#pragma unroll
        for (int ni = 0; ni < 4; ++ni)
            acc[mi][ni] = (v8i){0, 0, 0, 0, 0, 0, 0, 0};

    const unsigned char* aptr[2];
#pragma unroll
    for (int mi = 0; mi < 2; ++mi)
        aptr[mi] = xb + (size_t)(m_base + mi * 16 + l16) * L_DIM + half * 16;
    const unsigned char* bptr[4];
#pragma unroll
    for (int ni = 0; ni < 4; ++ni)
        bptr[ni] = rb + (size_t)(n_base + ni * 16 + l16) * L_DIM + half * 16;

    v8i a0[2], a1[2], b0[4], b1[4];
#pragma unroll
    for (int mi = 0; mi < 2; ++mi) a0[mi] = load_a_frag(aptr[mi]);
#pragma unroll
    for (int ni = 0; ni < 4; ++ni) b0[ni] = load_b_frag(bptr[ni]);

#pragma unroll 1
    for (int k0 = 0; k0 < L_DIM - 128; k0 += 128) {
#pragma unroll
        for (int mi = 0; mi < 2; ++mi) a1[mi] = load_a_frag(aptr[mi] + k0 + 64);
#pragma unroll
        for (int ni = 0; ni < 4; ++ni) b1[ni] = load_b_frag(bptr[ni] + k0 + 64);
#pragma unroll
        for (int mi = 0; mi < 2; ++mi)
#pragma unroll
            for (int ni = 0; ni < 4; ++ni)
                acc[mi][ni] = __builtin_amdgcn_wmma_i32_16x16x64_iu8(
                    false, a0[mi], false, b0[ni], acc[mi][ni], false, false);
#pragma unroll
        for (int mi = 0; mi < 2; ++mi) a0[mi] = load_a_frag(aptr[mi] + k0 + 128);
#pragma unroll
        for (int ni = 0; ni < 4; ++ni) b0[ni] = load_b_frag(bptr[ni] + k0 + 128);
#pragma unroll
        for (int mi = 0; mi < 2; ++mi)
#pragma unroll
            for (int ni = 0; ni < 4; ++ni)
                acc[mi][ni] = __builtin_amdgcn_wmma_i32_16x16x64_iu8(
                    false, a1[mi], false, b1[ni], acc[mi][ni], false, false);
    }

#pragma unroll
    for (int mi = 0; mi < 2; ++mi) a1[mi] = load_a_frag(aptr[mi] + (L_DIM - 64));
#pragma unroll
    for (int ni = 0; ni < 4; ++ni) b1[ni] = load_b_frag(bptr[ni] + (L_DIM - 64));
#pragma unroll
    for (int mi = 0; mi < 2; ++mi)
#pragma unroll
        for (int ni = 0; ni < 4; ++ni)
            acc[mi][ni] = __builtin_amdgcn_wmma_i32_16x16x64_iu8(
                false, a0[mi], false, b0[ni], acc[mi][ni], false, false);
#pragma unroll
    for (int mi = 0; mi < 2; ++mi)
#pragma unroll
        for (int ni = 0; ni < 4; ++ni)
            acc[mi][ni] = __builtin_amdgcn_wmma_i32_16x16x64_iu8(
                false, a1[mi], false, b1[ni], acc[mi][ni], false, false);

    const float invstd = 0.04419417382415922f;
    float sx[2][8];
#pragma unroll
    for (int mi = 0; mi < 2; ++mi)
#pragma unroll
        for (int r = 0; r < 8; ++r)
            sx[mi][r] = sum_x[m_base + mi * 16 + half * 8 + r];
    float sr[4];
#pragma unroll
    for (int ni = 0; ni < 4; ++ni)
        sr[ni] = sum_r[n_base + ni * 16 + l16];

#pragma unroll
    for (int mi = 0; mi < 2; ++mi)
#pragma unroll
        for (int r = 0; r < 8; ++r) {
            const int row = m_base + mi * 16 + half * 8 + r;
            float* orow = out + (size_t)row * D_MODEL;
#pragma unroll
            for (int ni = 0; ni < 4; ++ni) {
                const int col = n_base + ni * 16 + l16;
                const float dot = (float)acc[mi][ni][r];
                orow[col] = (sx[mi][r] + sr[ni] - 2.0f * dot - 1024.0f) * invstd;
            }
        }
}

extern "C" void kernel_launch(void* const* d_in, const int* in_sizes, int n_in,
                              void* d_out, int out_size, void* d_ws, size_t ws_size,
                              hipStream_t stream) {
    const int*   x    = (const int*)d_in[0];
    const float* refs = (const float*)d_in[1];
    float*       out  = (float*)d_out;

    unsigned char* ws = (unsigned char*)d_ws;
    unsigned char* xb = ws;
    unsigned char* rb = ws + (size_t)BATCH * L_DIM;
    float* sum_x = (float*)(rb + (size_t)D_MODEL * L_DIM);
    float* sum_r = sum_x + BATCH;

    pack_kernel<<<BATCH + D_MODEL, 256, 0, stream>>>(x, refs, xb, rb, sum_x, sum_r);

    dim3 grid(BATCH / 128, D_MODEL / 128);
    hamming_wmma_kernel<<<grid, 256, 0, stream>>>(xb, rb, sum_x, sum_r, out);
}
